// Deepfm_67491116089396
// MI455X (gfx1250) — matrix operand order repaired and checked by reading, not compiled, not run on hardware
//
#include <hip/hip_runtime.h>
#include <hip/hip_bf16.h>

typedef __attribute__((ext_vector_type(16))) _Float16 v16h;
typedef __attribute__((ext_vector_type(8)))  _Float16 v8h;
typedef __attribute__((ext_vector_type(8)))  float    v8f;

// Problem constants (from the reference)
#define B_N    16384
#define NS_N   26
#define ND_N   13
#define V_N    100000
#define E_N    16
#define H_N    400
#define K0PAD  448     // 429 padded up to multiple of 32
#define HPAD   416     // 400 padded up to multiple of 32
#define NTILES 25      // 400 / 16 output-column tiles

// ---------------------------------------------------------------------------
// Weight prep: W [K x N] row-major f32  ->  WT [N x Kpad] row-major f16
// (so a B-matrix fragment for one lane is two contiguous 8-half runs)
// ---------------------------------------------------------------------------
__global__ __launch_bounds__(256) void prep_w(const float* __restrict__ W,
                                              _Float16* __restrict__ WT,
                                              int K, int Kpad, int N) {
    int t = blockIdx.x * blockDim.x + threadIdx.x;
    int total = N * Kpad;
    if (t >= total) return;
    int n = t / Kpad;
    int k = t - n * Kpad;
    float v = (k < K) ? W[(size_t)k * N + n] : 0.0f;
    WT[(size_t)n * Kpad + k] = (_Float16)v;
}

// ---------------------------------------------------------------------------
// Gather + FM terms.  One wave32 per batch row.
//   act0[b][0..415]   = e2 flattened (feature-major), f16
//   act0[b][416..428] = X_dense, f16
//   act0[b][429..447] = 0
//   logit0[b] = sum(e1) + X_dense.lin_w + 0.5*sum(sos - ssq) + bias
// ---------------------------------------------------------------------------
__global__ __launch_bounds__(256) void gather_fm(const int*   __restrict__ Xs,
                                                 const float* __restrict__ Xd,
                                                 const float* __restrict__ emb1,
                                                 const float* __restrict__ emb2,
                                                 const float* __restrict__ linw,
                                                 const float* __restrict__ bias0,
                                                 _Float16*    __restrict__ act0,
                                                 float*       __restrict__ logit0) {
    int wid  = (blockIdx.x * blockDim.x + threadIdx.x) >> 5;
    int lane = threadIdx.x & 31;
    if (wid >= B_N) return;
    const int b = wid;
    const int* xs = Xs + (size_t)b * NS_N;

    float sum_d = 0.0f, sumsq = 0.0f;
    if (lane < 16) {
        #pragma unroll
        for (int f = 0; f < NS_N; ++f) {
            int idx = xs[f];
            float v = emb2[((size_t)f * V_N + (size_t)idx) * E_N + lane];
            sum_d += v;
            sumsq += v * v;
            act0[(size_t)b * K0PAD + f * E_N + lane] = (_Float16)v;
        }
    }
    // linear terms
    float lin = 0.0f;
    if (lane < NS_N) lin += emb1[(size_t)lane * V_N + (size_t)xs[lane]];
    if (lane < ND_N) {
        float xd = Xd[(size_t)b * ND_N + lane];
        lin += xd * linw[lane];
        act0[(size_t)b * K0PAD + 416 + lane] = (_Float16)xd;
    } else {
        // lanes 13..31 zero-fill pad cols 429..447
        act0[(size_t)b * K0PAD + 416 + lane] = (_Float16)0.0f;
    }
    float cr = (lane < 16) ? (sum_d * sum_d - sumsq) : 0.0f;
    #pragma unroll
    for (int off = 16; off > 0; off >>= 1) {
        cr  += __shfl_xor(cr,  off, 32);
        lin += __shfl_xor(lin, off, 32);
    }
    if (lane == 0) logit0[b] = lin + 0.5f * cr + bias0[0];
}

// ---------------------------------------------------------------------------
// A-fragment loader: 16-bit A matrix 16x32 per ISA layout.
// lanes 0-15 : K = k0+0..7 (elems 0-7), K = k0+16..23 (elems 8-15)
// lanes16-31 : K = k0+8..15,            K = k0+24..31
// p points at row_base + k0 + sel*8  -> two 16B contiguous runs.
// ---------------------------------------------------------------------------
__device__ __forceinline__ v16h load_afrag(const _Float16* __restrict__ p) {
    v8h lo = *(const v8h*)(p);
    v8h hi = *(const v8h*)(p + 16);
    v16h r;
    #pragma unroll
    for (int i = 0; i < 8; ++i) { r[i] = lo[i]; r[8 + i] = hi[i]; }
    return r;
}

// ---------------------------------------------------------------------------
// GEMM + bias + ReLU via v_wmma_f32_16x16x32_f16.
// Each wave: 64 rows x 16 cols (4 M-tiles reuse one B fragment per K step).
// act: [B x Kpad] f16 row-major, wt: [400 x Kpad] f16 (pre-transposed),
// out: [B x HPAD] f16, cols 400..415 zero-padded by the nt==24 waves.
// ---------------------------------------------------------------------------
__global__ __launch_bounds__(256) void gemm_relu_wmma(const _Float16* __restrict__ act,
                                                      const _Float16* __restrict__ wt,
                                                      const float*    __restrict__ bias,
                                                      _Float16*       __restrict__ out,
                                                      int Kpad) {
    int wid  = (blockIdx.x * blockDim.x + threadIdx.x) >> 5;
    int lane = threadIdx.x & 31;
    int nt = wid % NTILES;          // output column tile
    int mg = wid / NTILES;          // group of 4 M-tiles
    int tileN  = nt * 16;
    int tileM0 = mg * 64;
    int sel  = (lane >> 4) & 1;     // half-wave select
    int mrow = lane & 15;

    const _Float16* bptr = wt  + (size_t)(tileN + mrow) * Kpad + sel * 8;
    const _Float16* aptr = act + (size_t)(tileM0 + mrow) * Kpad + sel * 8;
    const size_t mstride = (size_t)16 * Kpad;

    v8f c0 = {}, c1 = {}, c2 = {}, c3 = {};
    for (int k0 = 0; k0 < Kpad; k0 += 32) {
        v16h bf = load_afrag(bptr + k0);               // B frag: two 16B runs, same K order as A
        v16h a0 = load_afrag(aptr + k0);
        v16h a1 = load_afrag(aptr + mstride + k0);
        v16h a2 = load_afrag(aptr + 2 * mstride + k0);
        v16h a3 = load_afrag(aptr + 3 * mstride + k0);
        c0 = __builtin_amdgcn_wmma_f32_16x16x32_f16(false, a0, false, bf, (short)0, c0, false, false);
        c1 = __builtin_amdgcn_wmma_f32_16x16x32_f16(false, a1, false, bf, (short)0, c1, false, false);
        c2 = __builtin_amdgcn_wmma_f32_16x16x32_f16(false, a2, false, bf, (short)0, c2, false, false);
        c3 = __builtin_amdgcn_wmma_f32_16x16x32_f16(false, a3, false, bf, (short)0, c3, false, false);
    }

    // Epilogue: D layout -> VGPR r holds M = r + sel*8, N = lane&15
    float bv = bias[tileN + mrow];
    v8f cc[4] = {c0, c1, c2, c3};
    #pragma unroll
    for (int mt = 0; mt < 4; ++mt) {
        size_t rowbase = (size_t)(tileM0 + mt * 16 + sel * 8);
        #pragma unroll
        for (int r = 0; r < 8; ++r) {
            float v = cc[mt][r] + bv;
            v = v > 0.0f ? v : 0.0f;
            out[(rowbase + r) * HPAD + tileN + mrow] = (_Float16)v;
        }
    }
    // Zero-pad cols 400..415 so next layer's K loop reads zeros.
    if (nt == NTILES - 1) {
        uint4 z = {0u, 0u, 0u, 0u};
        #pragma unroll
        for (int mt = 0; mt < 4; ++mt) {
            _Float16* pz = out + (size_t)(tileM0 + mt * 16 + mrow) * HPAD + 400 + sel * 8;
            *(uint4*)pz = z;   // 8 halves, 16B aligned
        }
    }
}

// ---------------------------------------------------------------------------
// Final: out[b] = logit0[b] + h2[b,:] . W_out
// ---------------------------------------------------------------------------
__global__ __launch_bounds__(256) void final_dot(const _Float16* __restrict__ h2,
                                                 const float*    __restrict__ Wout,
                                                 const float*    __restrict__ logit0,
                                                 float*          __restrict__ out) {
    int wid  = (blockIdx.x * blockDim.x + threadIdx.x) >> 5;
    int lane = threadIdx.x & 31;
    if (wid >= B_N) return;
    const int b = wid;
    float acc = 0.0f;
    for (int n = lane; n < H_N; n += 32)
        acc += (float)h2[(size_t)b * HPAD + n] * Wout[n];
    #pragma unroll
    for (int off = 16; off > 0; off >>= 1) acc += __shfl_xor(acc, off, 32);
    if (lane == 0) out[b] = logit0[b] + acc;
}

// ---------------------------------------------------------------------------
extern "C" void kernel_launch(void* const* d_in, const int* in_sizes, int n_in,
                              void* d_out, int out_size, void* d_ws, size_t ws_size,
                              hipStream_t stream) {
    (void)in_sizes; (void)n_in; (void)out_size; (void)ws_size;
    const int*   Xs    = (const int*)  d_in[0];   // [B, 26]
    const float* Xd    = (const float*)d_in[1];   // [B, 13]
    const float* emb1  = (const float*)d_in[2];   // [26, V, 1]
    const float* emb2  = (const float*)d_in[3];   // [26, V, 16]
    const float* linw  = (const float*)d_in[4];   // [13, 1]
    const float* bias0 = (const float*)d_in[5];   // [1]
    const float* W0    = (const float*)d_in[6];   // [429, 400]
    const float* b0    = (const float*)d_in[7];   // [400]
    const float* W1    = (const float*)d_in[8];   // [400, 400]
    const float* b1    = (const float*)d_in[9];   // [400]
    const float* W2    = (const float*)d_in[10];  // [400, 400]
    const float* b2    = (const float*)d_in[11];  // [400]
    const float* Wout  = (const float*)d_in[12];  // [400, 1]
    float* out = (float*)d_out;                   // [B, 1]

    // Workspace layout (256B-aligned offsets)
    char* ws = (char*)d_ws;
    size_t off = 0;
    _Float16* act0 = (_Float16*)(ws + off); off += (size_t)B_N * K0PAD * 2;  // 14,680,064
    _Float16* actA = (_Float16*)(ws + off); off += (size_t)B_N * HPAD * 2;   // 13,631,488
    _Float16* actB = (_Float16*)(ws + off); off += (size_t)B_N * HPAD * 2;   // 13,631,488
    float*   logit0 = (float*)(ws + off);   off += (size_t)B_N * 4;          // 65,536
    _Float16* W0h = (_Float16*)(ws + off);  off += (size_t)H_N * K0PAD * 2;  // 358,400
    _Float16* W1h = (_Float16*)(ws + off);  off += (size_t)H_N * HPAD * 2;   // 332,800
    _Float16* W2h = (_Float16*)(ws + off);  off += (size_t)H_N * HPAD * 2;   // 332,800

    // 1) weight conversion / transpose / pad
    {
        int t0 = H_N * K0PAD, t1 = H_N * HPAD;
        prep_w<<<(t0 + 255) / 256, 256, 0, stream>>>(W0, W0h, 429, K0PAD, H_N);
        prep_w<<<(t1 + 255) / 256, 256, 0, stream>>>(W1, W1h, 400, HPAD,  H_N);
        prep_w<<<(t1 + 255) / 256, 256, 0, stream>>>(W2, W2h, 400, HPAD,  H_N);
    }

    // 2) gather + FM terms (one wave per row, 8 waves/block)
    gather_fm<<<B_N / 8, 256, 0, stream>>>(Xs, Xd, emb1, emb2, linw, bias0, act0, logit0);

    // 3) three hidden layers via WMMA (6400 waves = 800 blocks each)
    const int gemm_blocks = (B_N / 64) * NTILES / 8;   // 256*25/8 = 800
    gemm_relu_wmma<<<gemm_blocks, 256, 0, stream>>>(act0, W0h, b0, actA, K0PAD);
    gemm_relu_wmma<<<gemm_blocks, 256, 0, stream>>>(actA, W1h, b1, actB, HPAD);
    gemm_relu_wmma<<<gemm_blocks, 256, 0, stream>>>(actB, W2h, b2, actA, HPAD);

    // 4) final dot + FM logit
    final_dot<<<B_N / 8, 256, 0, stream>>>(actA, Wout, logit0, out);
}
